// TransformerBlock_27977416966420
// MI455X (gfx1250) — matrix operand order repaired and checked by reading, not compiled, not run on hardware
//
#include <hip/hip_runtime.h>
#include <math.h>

// ---------------- common types / helpers ----------------
typedef __attribute__((ext_vector_type(16))) __bf16 v16bf;
typedef __attribute__((ext_vector_type(8)))  float  v8f;
typedef __attribute__((__vector_size__(16))) int    vi4;   // matches builtin param type

union FragB { v16bf b; unsigned int u[8]; };

__device__ inline unsigned short f2bf(float f) {
  unsigned int u = __builtin_bit_cast(unsigned int, f);
  u += 0x7FFFu + ((u >> 16) & 1u);            // round-to-nearest-even
  return (unsigned short)(u >> 16);
}
__device__ inline float bf2f(unsigned short s) {
  return __builtin_bit_cast(float, ((unsigned int)s) << 16);
}

constexpr int Bb = 2, Tt = 2048, Dd = 1024, Hh = 16, HDc = 64, Ee = 8, Ff = 2048;
constexpr int NT = Bb * Tt;                    // 4096 tokens

// -------- gfx1250 async global->LDS (guarded: compile-safe either way) ------
#if defined(__has_builtin)
#if __has_builtin(__builtin_amdgcn_global_load_async_to_lds_b128)
#define HAVE_ASYNC_LDS 1
#endif
#if __has_builtin(__builtin_amdgcn_s_wait_asynccnt)
#define HAVE_WAIT_ASYNC 1
#endif
#endif

__device__ inline void cp_async16(void* lds, const void* g) {
#if defined(HAVE_ASYNC_LDS)
  __builtin_amdgcn_global_load_async_to_lds_b128(
      (__attribute__((address_space(1))) vi4*)(unsigned long long)g,
      (__attribute__((address_space(3))) vi4*)(unsigned int)(unsigned long long)lds,
      0, 0);
#else
  (void)lds; (void)g;
#endif
}
__device__ inline void wait_async_lds() {
#if defined(HAVE_ASYNC_LDS)
#if defined(HAVE_WAIT_ASYNC)
  __builtin_amdgcn_s_wait_asynccnt(0);
#else
  asm volatile("s_wait_asynccnt 0x0" ::: "memory");
#endif
#endif
}

// ---------------- f32 -> bf16 conversion ----------------
__global__ __launch_bounds__(256) void cvt_f32_bf16(const float* __restrict__ in,
                                                    unsigned short* __restrict__ out,
                                                    long long n) {
  long long i = (long long)blockIdx.x * 256 + threadIdx.x;
  if (i < n) out[i] = f2bf(in[i]);
}

// ---------------- LayerNorm (f32 in, bf16 [+f32] out) ----------------
__global__ __launch_bounds__(256) void ln_kernel(const float* __restrict__ x,
                                                 const float* __restrict__ g,
                                                 const float* __restrict__ bta,
                                                 unsigned short* __restrict__ outb,
                                                 float* __restrict__ outf) {
  __shared__ float s1[256], s2[256];
  int n = blockIdx.x, tid = threadIdx.x;
  const float* row = x + (size_t)n * Dd;
  float a = 0.f, b2 = 0.f;
  for (int i = tid; i < Dd; i += 256) { float v = row[i]; a += v; b2 += v * v; }
  s1[tid] = a; s2[tid] = b2; __syncthreads();
  for (int s = 128; s > 0; s >>= 1) {
    if (tid < s) { s1[tid] += s1[tid + s]; s2[tid] += s2[tid + s]; }
    __syncthreads();
  }
  float mu = s1[0] * (1.0f / Dd);
  float var = s2[0] * (1.0f / Dd) - mu * mu;
  float r = rsqrtf(var + 1e-5f);
  for (int i = tid; i < Dd; i += 256) {
    float v = (row[i] - mu) * r * g[i] + bta[i];
    outb[(size_t)n * Dd + i] = f2bf(v);
    if (outf) outf[(size_t)n * Dd + i] = v;
  }
}

// ---------------- generic bf16 WMMA GEMM (templated epilogue) ----------------
// C[M,N] = A[M,K](bf16) * B[K,N](bf16) [+ bias] [+ residual]; batch = blockIdx.z.
// Block tile 128x128, 8 waves as 4(m) x 2(n); wave tile 32x64 = 8 wmma / k-step.
// Double-buffered LDS; A staged via async global->LDS when available.
template <bool HB, bool HR, bool OB>
__global__ __launch_bounds__(256) void gemm_bf16_wmma(
    const unsigned short* __restrict__ A, const unsigned short* __restrict__ B,
    const float* __restrict__ bias, const float* __restrict__ residual,
    float* __restrict__ Cf, unsigned short* __restrict__ Cb,
    int M, int N, int K,
    long long sAb, long long sBb, long long sCb, long long sBiasb) {
  __shared__ unsigned short As[2][128][32];   // [buf][m][k] raw rows (64B, 16B-aligned)
  __shared__ unsigned short Bs[2][128][34];   // [buf][n][k] transposed, padded
  const int tid = threadIdx.x;
  const int wv = tid >> 5, lane = tid & 31;
  const int hi = lane >> 4, ln = lane & 15;
  const int wm = wv & 3, wn = wv >> 2;
  const int bz = blockIdx.z;
  A += (long long)bz * sAb;
  B += (long long)bz * sBb;
  const long long cO = (long long)bz * sCb;
  const int m0 = blockIdx.y * 128, n0 = blockIdx.x * 128;

  v8f zf = {0.f,0.f,0.f,0.f,0.f,0.f,0.f,0.f};
  v8f acc[2][4] = {{zf, zf, zf, zf}, {zf, zf, zf, zf}};

  // ---- tile fetch helpers ----
  auto fetchA = [&](int k0, int buf) {          // 128x32 bf16 = 512 x 16B chunks
#pragma unroll
    for (int i = 0; i < 2; ++i) {
      int c = tid + i * 256;
      int row = c >> 2, q = c & 3;
      const void* g = A + (long long)(m0 + row) * K + k0 + q * 8;
#if defined(HAVE_ASYNC_LDS)
      cp_async16(&As[buf][row][q * 8], g);
#else
      *(uint4*)&As[buf][row][q * 8] = *(const uint4*)g;
#endif
    }
  };
  auto loadBreg = [&](int k0, unsigned* t) {    // 32x128 bf16 = 2048 dwords
#pragma unroll
    for (int i = 0; i < 8; ++i) {
      int didx = tid + i * 256;
      int kr = didx >> 6, dc = didx & 63;
      t[i] = *(const unsigned int*)(B + (long long)(k0 + kr) * N + n0 + dc * 2);
    }
  };
  auto storeBreg = [&](const unsigned* t, int buf) {
#pragma unroll
    for (int i = 0; i < 8; ++i) {
      int didx = tid + i * 256;
      int kr = didx >> 6, dc = didx & 63;
      Bs[buf][dc * 2][kr]     = (unsigned short)(t[i] & 0xFFFFu);
      Bs[buf][dc * 2 + 1][kr] = (unsigned short)(t[i] >> 16);
    }
  };

  // ---- prologue: stage tile k0=0 into buffer 0 ----
  {
    unsigned bt[8];
    fetchA(0, 0);
    loadBreg(0, bt);
    storeBreg(bt, 0);
    wait_async_lds();
    __syncthreads();
  }

  int cur = 0;
  for (int k0 = 0; k0 < K; k0 += 32) {
    const bool hasNext = (k0 + 32 < K);
    const int nxt = cur ^ 1;
    unsigned bt[8];
    if (hasNext) {
      fetchA(k0 + 32, nxt);                    // async: lands in LDS directly
      loadBreg(k0 + 32, bt);                   // sync: regs now, LDS after compute
    }
    if (k0 + 64 < K) {                         // prefetch tile after next into L2
      __builtin_prefetch(A + (long long)(m0 + (tid >> 1)) * K + k0 + 64, 0, 3);
      __builtin_prefetch(B + (long long)(k0 + 64 + (tid >> 5)) * N + n0 + (tid & 31) * 4, 0, 3);
    }

    // ---- compute on current buffer: gather all fragments, then 8 wmma ----
    FragB af[2];
#pragma unroll
    for (int r = 0; r < 2; ++r)
#pragma unroll
      for (int v = 0; v < 8; ++v) {
        int kk = 2 * v + 8 * hi + ((v >= 4) ? 8 : 0);   // 16-bit A layout
        af[r].u[v] = *(const unsigned int*)&As[cur][wm * 32 + r * 16 + ln][kk];
      }
    FragB bf4[4];
#pragma unroll
    for (int s = 0; s < 4; ++s)
#pragma unroll
      for (int v = 0; v < 8; ++v)                        // 16-bit B layout
        bf4[s].u[v] = *(const unsigned int*)&Bs[cur][wn * 64 + s * 16 + ln][2 * v + 8 * hi + ((v >= 4) ? 8 : 0)];
#pragma unroll
    for (int r = 0; r < 2; ++r)
#pragma unroll
      for (int s = 0; s < 4; ++s)
        acc[r][s] = __builtin_amdgcn_wmma_f32_16x16x32_bf16(
            false, af[r].b, false, bf4[s].b, (short)0, acc[r][s], false, false);

    if (hasNext) storeBreg(bt, nxt);
    wait_async_lds();
    __syncthreads();
    cur = nxt;
  }

  // ---- branch-free epilogue ----
  const float* biasP = HB ? (bias + (long long)bz * sBiasb) : nullptr;
#pragma unroll
  for (int r = 0; r < 2; ++r) {
#pragma unroll
    for (int s = 0; s < 4; ++s) {
      const int col = n0 + wn * 64 + s * 16 + ln;
      float bv = HB ? biasP[col] : 0.f;
#pragma unroll
      for (int i = 0; i < 8; ++i) {
        int row = m0 + wm * 32 + r * 16 + i + 8 * hi;   // C layout: VGPR i -> M=i+8*hi
        float val = acc[r][s][i] + bv;
        if (HR) val += residual[(long long)row * N + col];
        long long o = cO + (long long)row * N + col;
        if (OB) Cb[o] = f2bf(val);
        else    Cf[o] = val;
      }
    }
  }
}

// ---------------- RoPE + reshape + Q scaling ----------------
__global__ __launch_bounds__(64) void rope_kernel(
    const float* __restrict__ qkv, const int* __restrict__ posOff,
    unsigned short* __restrict__ Qr, unsigned short* __restrict__ Kr,
    unsigned short* __restrict__ Vr) {
  int n = blockIdx.x, h = blockIdx.y, d = threadIdx.x;
  int t = n & (Tt - 1), b = n / Tt;
  const float* row = qkv + (size_t)n * (3 * Hh * HDc);
  int i = d & 31;
  float pos = (float)(t + posOff[0]);
  float invf = powf(10000.0f, -((float)(2 * i)) * (1.0f / 64.0f));
  float ang = pos * invf;
  float cs = cosf(ang), sn = sinf(ang);
  float q1 = row[h * 64 + i],        q2 = row[h * 64 + i + 32];
  float k1 = row[1024 + h * 64 + i], k2 = row[1024 + h * 64 + i + 32];
  float vv = row[2048 + h * 64 + d];
  float qo = (d < 32) ? (q1 * cs - q2 * sn) : (q1 * sn + q2 * cs);
  float ko = (d < 32) ? (k1 * cs - k2 * sn) : (k1 * sn + k2 * cs);
  size_t idx = ((size_t)(b * Hh + h) * Tt + t) * 64 + d;
  Qr[idx] = f2bf(qo * 0.125f);   // 1/sqrt(HD) folded into Q
  Kr[idx] = f2bf(ko);
  Vr[idx] = f2bf(vv);
}

// ---------------- flash attention: 1 wave = 16 query rows ----------------
__global__ __launch_bounds__(32) void attn_kernel(
    const unsigned short* __restrict__ Q, const unsigned short* __restrict__ Kx,
    const unsigned short* __restrict__ V, unsigned short* __restrict__ O) {
  __shared__ unsigned short Ks[32][66];   // [key][d]
  __shared__ unsigned short Vs[64][34];   // [d][key] (transposed)
  __shared__ unsigned short Ps[16][32];   // probs restage for A fragment
  const int lane = threadIdx.x;
  const int hi = lane >> 4, ln = lane & 15;
  const int q0 = blockIdx.x * 16;
  const int bh = blockIdx.y, b = bh >> 4, h = bh & 15;

  FragB aq0, aq1;
  {
    const unsigned short* qrow = Q + ((size_t)bh * Tt + q0 + ln) * 64;
#pragma unroll
    for (int v = 0; v < 8; ++v) {
      int kk = 2 * v + 8 * hi + ((v >= 4) ? 8 : 0);
      aq0.u[v] = *(const unsigned int*)(qrow + kk);
      aq1.u[v] = *(const unsigned int*)(qrow + 32 + kk);
    }
  }

  v8f zf = {0.f,0.f,0.f,0.f,0.f,0.f,0.f,0.f};
  float m[8], l[8];
  v8f o[4];
#pragma unroll
  for (int i = 0; i < 8; ++i) { m[i] = -__builtin_inff(); l[i] = 0.f; }
#pragma unroll
  for (int t = 0; t < 4; ++t) o[t] = zf;

  const int nkt = ((q0 + 15) >> 5) + 1;
  for (int kt = 0; kt < nkt; ++kt) {
    const int k0 = kt * 32;
#pragma unroll
    for (int i = 0; i < 32; ++i) {     // stage 32 keys x 64 dims of K and V^T
      int didx = lane + i * 32;
      int key = didx >> 5, dc = didx & 31;
      unsigned int wk = *(const unsigned int*)(Kx + ((size_t)bh * Tt + k0 + key) * 64 + dc * 2);
      *(unsigned int*)&Ks[key][dc * 2] = wk;
      unsigned int wv = *(const unsigned int*)(V + ((size_t)bh * Tt + k0 + key) * 64 + dc * 2);
      Vs[dc * 2][key]     = (unsigned short)(wv & 0xFFFFu);
      Vs[dc * 2 + 1][key] = (unsigned short)(wv >> 16);
    }
    __syncthreads();

    v8f c0 = zf, c1 = zf;              // scores: keys [k0,k0+16) and [k0+16,k0+32)
#pragma unroll
    for (int t = 0; t < 2; ++t) {      // t = d slice (0..31 / 32..63)
      FragB bk0, bk1;
#pragma unroll
      for (int v = 0; v < 8; ++v) {
        int kk = 2 * v + 8 * hi + ((v >= 4) ? 8 : 0) + 32 * t;
        bk0.u[v] = *(const unsigned int*)&Ks[ln][kk];
        bk1.u[v] = *(const unsigned int*)&Ks[16 + ln][kk];
      }
      const v16bf aT = (t == 0) ? aq0.b : aq1.b;
      c0 = __builtin_amdgcn_wmma_f32_16x16x32_bf16(false, aT, false, bk0.b, (short)0, c0, false, false);
      c1 = __builtin_amdgcn_wmma_f32_16x16x32_bf16(false, aT, false, bk1.b, (short)0, c1, false, false);
    }

    // causal mask + online softmax (row reduction across a 16-lane half)
#pragma unroll
    for (int i = 0; i < 8; ++i) {
      int qI = q0 + i + 8 * hi;
      float v0 = c0[i]; if (k0 + ln > qI)      v0 = -3.0e38f;
      float v1 = c1[i]; if (k0 + 16 + ln > qI) v1 = -3.0e38f;
      float rm = fmaxf(v0, v1);
#pragma unroll
      for (int ofs = 8; ofs >= 1; ofs >>= 1) rm = fmaxf(rm, __shfl_xor(rm, ofs, 32));
      float nm = fmaxf(m[i], rm);
      float sc = expf(m[i] - nm);
      float p0 = expf(v0 - nm), p1 = expf(v1 - nm);
      float rs = p0 + p1;
#pragma unroll
      for (int ofs = 8; ofs >= 1; ofs >>= 1) rs += __shfl_xor(rs, ofs, 32);
      l[i] = l[i] * sc + rs;
      m[i] = nm;
#pragma unroll
      for (int t = 0; t < 4; ++t) o[t][i] = o[t][i] * sc;
      Ps[i + 8 * hi][ln]      = f2bf(p0);
      Ps[i + 8 * hi][16 + ln] = f2bf(p1);
    }
    __syncthreads();

    FragB ap;                          // P re-enters as a 16x32 A fragment
#pragma unroll
    for (int v = 0; v < 8; ++v) {
      int kk = 2 * v + 8 * hi + ((v >= 4) ? 8 : 0);
      ap.u[v] = *(const unsigned int*)&Ps[ln][kk];
    }
#pragma unroll
    for (int t = 0; t < 4; ++t) {      // O += P(16x32) * V(32x16) per d-subtile
      FragB bv;
#pragma unroll
      for (int v = 0; v < 8; ++v)
        bv.u[v] = *(const unsigned int*)&Vs[t * 16 + ln][2 * v + 8 * hi + ((v >= 4) ? 8 : 0)];
      o[t] = __builtin_amdgcn_wmma_f32_16x16x32_bf16(false, ap.b, false, bv.b, (short)0, o[t], false, false);
    }
    __syncthreads();
  }

#pragma unroll
  for (int t = 0; t < 4; ++t) {
#pragma unroll
    for (int i = 0; i < 8; ++i) {
      int tq = q0 + i + 8 * hi;
      int d = t * 16 + ln;
      float val = o[t][i] / l[i];
      O[((size_t)(b * Tt + tq)) * (Hh * HDc) + h * 64 + d] = f2bf(val);
    }
  }
}

// ---------------- gate: logits -> softmax(8) + utilization ----------------
__global__ __launch_bounds__(256) void gate_kernel(
    const float* __restrict__ h2, const float* __restrict__ gw,
    const float* __restrict__ gb, float* __restrict__ gp, float* __restrict__ util) {
  __shared__ float sh[8][256];
  int n = blockIdx.x, tid = threadIdx.x;
  float part[8];
#pragma unroll
  for (int e = 0; e < 8; ++e) part[e] = 0.f;
  const float* row = h2 + (size_t)n * Dd;
  for (int dd = tid; dd < Dd; dd += 256) {
    float hv = row[dd];
#pragma unroll
    for (int e = 0; e < 8; ++e) part[e] += hv * gw[dd * 8 + e];
  }
#pragma unroll
  for (int e = 0; e < 8; ++e) sh[e][tid] = part[e];
  __syncthreads();
  for (int s = 128; s > 0; s >>= 1) {
    if (tid < s) {
#pragma unroll
      for (int e = 0; e < 8; ++e) sh[e][tid] += sh[e][tid + s];
    }
    __syncthreads();
  }
  if (tid == 0) {
    float lg[8], mx = -3.0e38f;
#pragma unroll
    for (int e = 0; e < 8; ++e) { lg[e] = sh[e][0] + gb[e]; mx = fmaxf(mx, lg[e]); }
    float sum = 0.f;
#pragma unroll
    for (int e = 0; e < 8; ++e) { lg[e] = expf(lg[e] - mx); sum += lg[e]; }
    float inv = 1.0f / sum;
#pragma unroll
    for (int e = 0; e < 8; ++e) {
      float p = lg[e] * inv;
      gp[(size_t)n * 8 + e] = p;
      atomicAdd(&util[e], p * (1.0f / (float)NT));
    }
  }
}

// ---------------- glu activation, gate-prescaled ----------------
// fc layout [e][n][2F]; actg layout [n][e*F + f] (matches eout_w flattened K)
__global__ __launch_bounds__(256) void act_kernel(
    const unsigned short* __restrict__ fc, const float* __restrict__ gp,
    unsigned short* __restrict__ actg) {
  long long idx = (long long)blockIdx.x * 256 + threadIdx.x;   // < NT*8*2048
  int f = (int)(idx & 2047);
  int e = (int)((idx >> 11) & 7);
  int n = (int)(idx >> 14);
  const unsigned short* base = fc + ((long long)e * NT + n) * (2 * Ff);
  float x1 = bf2f(base[f]);
  float x2 = bf2f(base[Ff + f]);
  float g = 0.5f * x2 * (1.0f + erff(x2 * 0.70710678118654752f));  // exact gelu
  float a = x1 * g * gp[(size_t)n * 8 + e];
  actg[(long long)n * (Ee * Ff) + e * Ff + f] = f2bf(a);
}

// ---------------- residual2 = xattn + sum_e gate*eout_b ----------------
__global__ __launch_bounds__(256) void res2_kernel(
    const float* __restrict__ xattn, const float* __restrict__ gp,
    const float* __restrict__ eob, float* __restrict__ out) {
  __shared__ float g8[8];
  int n = blockIdx.x, tid = threadIdx.x;
  if (tid < 8) g8[tid] = gp[(size_t)n * 8 + tid];
  __syncthreads();
  for (int d = tid; d < Dd; d += 256) {
    float acc = xattn[(size_t)n * Dd + d];
#pragma unroll
    for (int e = 0; e < 8; ++e) acc += g8[e] * eob[e * Dd + d];
    out[(size_t)n * Dd + d] = acc;
  }
}

__global__ void zero8_kernel(float* p) { if (threadIdx.x < 8) p[threadIdx.x] = 0.f; }

__global__ void util_loss_kernel(const float* __restrict__ util, float* __restrict__ out) {
  if (threadIdx.x == 0) {
    float s = 0.f;
#pragma unroll
    for (int e = 0; e < 8; ++e) { float d = util[e] - 0.125f; s += d * d; }
    out[0] = s;
  }
}

// ---------------- host-side orchestration ----------------
extern "C" void kernel_launch(void* const* d_in, const int* in_sizes, int n_in,
                              void* d_out, int out_size, void* d_ws, size_t ws_size,
                              hipStream_t stream) {
  (void)in_sizes; (void)n_in; (void)out_size; (void)ws_size;
  const float* x    = (const float*)d_in[0];
  const int*   posO = (const int*)  d_in[2];     // d_in[1] = mask (computed analytically)
  const float* ln1g = (const float*)d_in[3];
  const float* ln1b = (const float*)d_in[4];
  const float* qkvw = (const float*)d_in[5];
  const float* qkvb = (const float*)d_in[6];
  const float* aow  = (const float*)d_in[7];
  const float* aob  = (const float*)d_in[8];
  const float* ln2g = (const float*)d_in[9];
  const float* ln2b = (const float*)d_in[10];
  const float* gw   = (const float*)d_in[11];
  const float* gb   = (const float*)d_in[12];
  const float* fcw  = (const float*)d_in[13];
  const float* fcb  = (const float*)d_in[14];
  const float* eow  = (const float*)d_in[15];
  const float* eob  = (const float*)d_in[16];
  float* out = (float*)d_out;

  char* ws = (char*)d_ws;
  size_t off = 0;
  auto alloc = [&](size_t bytes) -> char* {
    char* p = ws + off;
    off = (off + bytes + 255) & ~(size_t)255;
    return p;
  };

  unsigned short* hB    = (unsigned short*)alloc((size_t)NT * Dd * 2);
  unsigned short* wqkvB = (unsigned short*)alloc((size_t)Dd * 3072 * 2);
  unsigned short* waoB  = (unsigned short*)alloc((size_t)Dd * Dd * 2);
  unsigned short* wfcB  = (unsigned short*)alloc((size_t)Ee * Dd * 2 * Ff * 2);
  unsigned short* weoB  = (unsigned short*)alloc((size_t)Ee * Ff * Dd * 2);
  float*          qkvF  = (float*)         alloc((size_t)NT * 3072 * 4);
  unsigned short* Qr    = (unsigned short*)alloc((size_t)NT * Dd * 2);
  unsigned short* Kr    = (unsigned short*)alloc((size_t)NT * Dd * 2);
  unsigned short* Vr    = (unsigned short*)alloc((size_t)NT * Dd * 2);
  unsigned short* Ob    = (unsigned short*)alloc((size_t)NT * Dd * 2);
  float*          xatt  = (float*)         alloc((size_t)NT * Dd * 4);
  float*          h2f   = (float*)         alloc((size_t)NT * Dd * 4);
  unsigned short* h2b   = (unsigned short*)alloc((size_t)NT * Dd * 2);
  float*          gp    = (float*)         alloc((size_t)NT * 8 * 4);
  float*          util  = (float*)         alloc(8 * 4);
  unsigned short* fcB   = (unsigned short*)alloc((size_t)Ee * NT * 2 * Ff * 2);
  unsigned short* actg  = (unsigned short*)alloc((size_t)NT * Ee * Ff * 2);

  // 1. weights -> bf16
  auto cvt = [&](const float* src, unsigned short* dst, long long n) {
    cvt_f32_bf16<<<(unsigned)((n + 255) / 256), 256, 0, stream>>>(src, dst, n);
  };
  cvt(qkvw, wqkvB, (long long)Dd * 3072);
  cvt(aow,  waoB,  (long long)Dd * Dd);
  cvt(fcw,  wfcB,  (long long)Ee * Dd * 2 * Ff);
  cvt(eow,  weoB,  (long long)Ee * Ff * Dd);

  // 2. LN1 -> h (bf16)
  ln_kernel<<<NT, 256, 0, stream>>>(x, ln1g, ln1b, hB, nullptr);

  // 3. QKV GEMM (bias, no residual, f32 out)
  gemm_bf16_wmma<true, false, false><<<dim3(3072 / 128, NT / 128, 1), 256, 0, stream>>>(
      hB, wqkvB, qkvb, nullptr, qkvF, nullptr, NT, 3072, Dd, 0, 0, 0, 0);

  // 4. RoPE + reshape to [b,h,t,d] bf16 (Q pre-scaled)
  rope_kernel<<<dim3(NT, Hh), 64, 0, stream>>>(qkvF, posO, Qr, Kr, Vr);

  // 5. flash attention -> O bf16 in [b,t,h*d] layout
  attn_kernel<<<dim3(Tt / 16, Bb * Hh), 32, 0, stream>>>(Qr, Kr, Vr, Ob);

  // 6. attn_out GEMM (bias + residual x, f32 out) -> xattn
  gemm_bf16_wmma<true, true, false><<<dim3(Dd / 128, NT / 128, 1), 256, 0, stream>>>(
      Ob, waoB, aob, x, xatt, nullptr, NT, Dd, Dd, 0, 0, 0, 0);

  // 7. LN2 -> h2 (f32 + bf16)
  ln_kernel<<<NT, 256, 0, stream>>>(xatt, ln2g, ln2b, h2b, h2f);

  // 8. gate softmax + utilization
  zero8_kernel<<<1, 32, 0, stream>>>(util);
  gate_kernel<<<NT, 256, 0, stream>>>(h2f, gw, gb, gp, util);

  // 9. expert fc GEMMs (batched over 8 experts; bias, bf16 out)
  gemm_bf16_wmma<true, false, true><<<dim3(2 * Ff / 128, NT / 128, Ee), 256, 0, stream>>>(
      h2b, wfcB, fcb, nullptr, nullptr, fcB, NT, 2 * Ff, Dd,
      0, (long long)Dd * 2 * Ff, (long long)NT * 2 * Ff, 2 * Ff);

  // 10. glu activation, pre-scaled by gate prob -> actg (N x 16384 bf16)
  act_kernel<<<(unsigned)(((long long)NT * Ee * Ff) / 256), 256, 0, stream>>>(fcB, gp, actg);

  // 11. residual2 = xattn + sum_e gate*eout_b  (into d_out)
  res2_kernel<<<NT, 256, 0, stream>>>(xatt, gp, eob, out);

  // 12. combined expert-out GEMM: (N x 16384) @ (16384 x D) + residual -> d_out
  gemm_bf16_wmma<false, true, false><<<dim3(Dd / 128, NT / 128, 1), 256, 0, stream>>>(
      actg, weoB, nullptr, out, out, nullptr, NT, Dd, Ee * Ff, 0, 0, 0, 0);

  // 13. utilization loss scalar
  util_loss_kernel<<<1, 32, 0, stream>>>(util, out + (size_t)NT * Dd);
}
